// TransformerBlock_66941360275827
// MI455X (gfx1250) — matrix operand order repaired and checked by reading, not compiled, not run on hardware
//
#include <hip/hip_runtime.h>
#include <hip/hip_bf16.h>
#include <math.h>

// ---------------------------------------------------------------------------
// CrossFormer-style block on gfx1250 (MI455X), wave32 + WMMA f16 pipeline.
// GEMMs run through v_wmma_f32_16x16x32_f16; the GEMM B-tile is fed by the
// Tensor Data Mover (tensor_load_to_lds + s_wait_tensorcnt) when available.
// Weights are pre-transposed to N-major once so all LDS staging is b128.
// ---------------------------------------------------------------------------

typedef __attribute__((ext_vector_type(16))) _Float16 v16h;
typedef __attribute__((ext_vector_type(8)))  float    v8f;
typedef __attribute__((ext_vector_type(8)))  _Float16 h8;
typedef __attribute__((ext_vector_type(4)))  unsigned u32x4;
typedef __attribute__((ext_vector_type(8)))  int      i32x8;
typedef __attribute__((ext_vector_type(4)))  int      i32x4;

#define TOKENS   65536   // 256*256
#define DIM      192
#define QKV_DIM  576
#define HIDDEN   768
#define NHEADS   6
#define HEADD    32

__device__ __forceinline__ float gelu_exact(float v) {
    return 0.5f * v * (1.0f + erff(v * 0.70710678118654752f));
}

// ---------------------------------------------------------------------------
// f32 (K x N) -> f16 transposed (N x K) weight conversion
// ---------------------------------------------------------------------------
__global__ void convert_transpose(const float* __restrict__ src,
                                  _Float16* __restrict__ dst, int K, int N) {
    int n = blockIdx.x * 16 + threadIdx.x;
    int k = blockIdx.y * 16 + threadIdx.y;
    if (k < K && n < N) dst[(size_t)n * K + k] = (_Float16)src[(size_t)k * N + n];
}

// ---------------------------------------------------------------------------
// LayerNorm over C=192, one wave (32 lanes) per token, 8 waves per block.
// ---------------------------------------------------------------------------
__global__ void ln_kernel(const float* __restrict__ x,
                          const float* __restrict__ g,
                          const float* __restrict__ b,
                          _Float16* __restrict__ out) {
    int wid  = threadIdx.x >> 5;
    int lane = threadIdx.x & 31;
    size_t tok = (size_t)blockIdx.x * 8 + wid;
    const float* row = x + tok * DIM;

    float vals[6];
    float s = 0.f;
#pragma unroll
    for (int i = 0; i < 6; ++i) { vals[i] = row[lane + i * 32]; s += vals[i]; }
#pragma unroll
    for (int off = 16; off > 0; off >>= 1) s += __shfl_xor(s, off, 32);
    float mu = s * (1.0f / DIM);

    float vs = 0.f;
#pragma unroll
    for (int i = 0; i < 6; ++i) { float d = vals[i] - mu; vs += d * d; }
#pragma unroll
    for (int off = 16; off > 0; off >>= 1) vs += __shfl_xor(vs, off, 32);
    float inv = rsqrtf(vs * (1.0f / DIM) + 1e-5f);

    _Float16* orow = out + tok * DIM;
#pragma unroll
    for (int i = 0; i < 6; ++i) {
        int c = lane + i * 32;
        orow[c] = (_Float16)((vals[i] - mu) * inv * g[c] + b[c]);
    }
}

// ---------------------------------------------------------------------------
// DynamicPosBias MLP: one thread per relative-offset row.
// ---------------------------------------------------------------------------
__device__ __forceinline__ void ln12_relu(const float* __restrict__ in,
                                          float* __restrict__ out,
                                          const float* g, const float* b) {
    float mu = 0.f;
#pragma unroll
    for (int j = 0; j < 12; ++j) mu += in[j];
    mu *= (1.0f / 12.0f);
    float var = 0.f;
#pragma unroll
    for (int j = 0; j < 12; ++j) { float d = in[j] - mu; var += d * d; }
    var *= (1.0f / 12.0f);
    float inv = rsqrtf(var + 1e-5f);
#pragma unroll
    for (int j = 0; j < 12; ++j) {
        float t = (in[j] - mu) * inv * g[j] + b[j];
        out[j] = t > 0.f ? t : 0.f;
    }
}

__global__ void posmlp_kernel(const float* w0, const float* b0,
                              const float* ln1g, const float* ln1b,
                              const float* w1, const float* b1,
                              const float* ln2g, const float* ln2b,
                              const float* w2, const float* b2,
                              const float* ln3g, const float* ln3b,
                              const float* w3, const float* b3,
                              float* __restrict__ posD,
                              float* __restrict__ posS) {
    int i = blockIdx.x * blockDim.x + threadIdx.x;
    float dh, dw; float* out;
    if (i < 225) { dh = (float)(i / 15 - 7);  dw = (float)(i % 15 - 7);  out = posD + i * 6; }
    else if (i < 225 + 961) { int j = i - 225; dh = (float)(j / 31 - 15); dw = (float)(j % 31 - 15); out = posS + j * 6; }
    else return;

    float h[12], t[12];
#pragma unroll
    for (int j = 0; j < 12; ++j) h[j] = dh * w0[j] + dw * w0[12 + j] + b0[j];

    ln12_relu(h, t, ln1g, ln1b);
#pragma unroll
    for (int o = 0; o < 12; ++o) {
        float a = b1[o];
#pragma unroll
        for (int j = 0; j < 12; ++j) a += t[j] * w1[j * 12 + o];
        h[o] = a;
    }
    ln12_relu(h, t, ln2g, ln2b);
#pragma unroll
    for (int o = 0; o < 12; ++o) {
        float a = b2[o];
#pragma unroll
        for (int j = 0; j < 12; ++j) a += t[j] * w2[j * 12 + o];
        h[o] = a;
    }
    ln12_relu(h, t, ln3g, ln3b);
#pragma unroll
    for (int o = 0; o < 6; ++o) {
        float a = b3[o];
#pragma unroll
        for (int j = 0; j < 12; ++j) a += t[j] * w3[j * 6 + o];
        out[o] = a;
    }
}

// ---------------------------------------------------------------------------
// TDM helper: issue a 2D tile load (tileK x tileN halves) from a N-major
// f16 matrix (row stride K halves) into LDS at ldsAddr. One call per wave.
// ---------------------------------------------------------------------------
__device__ __forceinline__ void tdm_load_tile_2d(const _Float16* gtile,
                                                 unsigned ldsAddr,
                                                 int K, int N,
                                                 int tileK, int tileN) {
#if __has_builtin(__builtin_amdgcn_tensor_load_to_lds)
    unsigned long long ga = (unsigned long long)(__UINTPTR_TYPE__)gtile;
    u32x4 g0;
    g0.x = 1u;                                        // count=1, user mode
    g0.y = ldsAddr;                                   // lds_addr (bytes)
    g0.z = (unsigned)ga;                              // global_addr[31:0]
    g0.w = (unsigned)((ga >> 32) & 0x01FFFFFFull)     // global_addr[56:32]
         | (2u << 30);                                // type = 2 (image)
    i32x8 g1;
    g1[0] = (1 << 16);                                // data_size=1 -> 2 bytes
    g1[1] = (int)((unsigned)(K & 0xffff) << 16);      // tensor_dim0[15:0] @bit48
    g1[2] = (int)(((unsigned)K >> 16) & 0xffffu)      // tensor_dim0[31:16]
          | (int)((unsigned)(N & 0xffff) << 16);      // tensor_dim1[15:0] @bit80
    g1[3] = (int)(((unsigned)N >> 16) & 0xffffu)      // tensor_dim1[31:16]
          | (int)((unsigned)tileK << 16);             // tile_dim0 @bit112
    g1[4] = tileN;                                    // tile_dim1 (tile_dim2=0)
    g1[5] = K;                                        // tensor_dim0_stride[31:0]
    g1[6] = 0;
    g1[7] = 0;
    i32x4 gz = {0, 0, 0, 0};
#if defined(__clang_major__) && (__clang_major__ >= 23)
    i32x8 gz8 = {0, 0, 0, 0, 0, 0, 0, 0};
    __builtin_amdgcn_tensor_load_to_lds(g0, g1, gz, gz, gz8, 0);
#else
    __builtin_amdgcn_tensor_load_to_lds(g0, g1, gz, gz, 0);
#endif
#else
    (void)gtile; (void)ldsAddr; (void)K; (void)N; (void)tileK; (void)tileN;
#endif
}

// ---------------------------------------------------------------------------
// WMMA GEMM: C[M,N] = act(A[M,K](f16) * BT[N,K](f16)^T + bias) (+res)
// 256 threads = 8 waves; block tile 128x64; wave tile 32x32 (4 accumulators).
// B tile delivered by the Tensor Data Mover (async) when available.
// ---------------------------------------------------------------------------
__global__ __launch_bounds__(256)
void gemm_wmma(const _Float16* __restrict__ A, int lda,
               const _Float16* __restrict__ BT,      // N-major, ld = K
               const float* __restrict__ bias,
               void* __restrict__ Cout, int ldc, int coff,
               const float* __restrict__ res, int ldr,
               int M, int N, int K, int act, int outHalf) {
    __shared__ _Float16 As[128][32];
    __shared__ _Float16 Bt[64][32];   // N-major tile: Bt[n][k]

    int tid   = threadIdx.x;
    int wid   = tid >> 5;
    int lane  = tid & 31;
    int khalf = lane >> 4;
    int m0 = blockIdx.y * 128, n0 = blockIdx.x * 64;
    int wr = wid >> 1, wc = wid & 1;

    v8f acc[2][2] = {};

    int a_r  = tid >> 1, a_kp = (tid & 1) * 16;
#if !__has_builtin(__builtin_amdgcn_tensor_load_to_lds)
    int b_r  = tid >> 2, b_kp = (tid & 3) * 8;
#endif
    unsigned ldsB = (unsigned)(__UINTPTR_TYPE__)&Bt[0][0];

    for (int k0 = 0; k0 < K; k0 += 32) {
#if __has_builtin(__builtin_amdgcn_tensor_load_to_lds)
        if (wid == 0)
            tdm_load_tile_2d(BT + (size_t)n0 * K + k0, ldsB, K, N, 32, 64);
#else
        *(h8*)&Bt[b_r][b_kp] =
            *(const h8*)(BT + (size_t)(n0 + b_r) * K + k0 + b_kp);
#endif
        // Stage A tile (128 x 32): 16 halves per thread, two b128 ops
        const _Float16* arow = A + (size_t)(m0 + a_r) * lda + k0 + a_kp;
        *(h8*)&As[a_r][a_kp]     = *(const h8*)(arow);
        *(h8*)&As[a_r][a_kp + 8] = *(const h8*)(arow + 8);
        if (k0 + 32 < K) __builtin_prefetch(arow + 32, 0, 1);

#if __has_builtin(__builtin_amdgcn_tensor_load_to_lds)
        if (wid == 0) __builtin_amdgcn_s_wait_tensorcnt(0);
#endif
        __syncthreads();

        union { v16h v; unsigned u[8]; } af[2], bf[2];
#pragma unroll
        for (int mt = 0; mt < 2; ++mt) {
            int mrow = wr * 32 + mt * 16 + (lane & 15);
#pragma unroll
            for (int v = 0; v < 8; ++v) {
                int k = (v >> 2) * 16 + khalf * 8 + (v & 3) * 2;
                af[mt].u[v] = *(const unsigned*)&As[mrow][k];
            }
        }
#pragma unroll
        for (int nt = 0; nt < 2; ++nt) {
            int nrow = wc * 32 + nt * 16 + (lane & 15);
#pragma unroll
            for (int v = 0; v < 8; ++v) {
                int k = (v >> 2) * 16 + khalf * 8 + (v & 3) * 2;
                bf[nt].u[v] = *(const unsigned*)&Bt[nrow][k];
            }
        }
#pragma unroll
        for (int mt = 0; mt < 2; ++mt)
#pragma unroll
            for (int nt = 0; nt < 2; ++nt)
                acc[mt][nt] = __builtin_amdgcn_wmma_f32_16x16x32_f16(
                    false, af[mt].v, false, bf[nt].v, (short)0,
                    acc[mt][nt], false, false);
        __syncthreads();
    }

    // Epilogue
#pragma unroll
    for (int mt = 0; mt < 2; ++mt) {
#pragma unroll
        for (int v = 0; v < 8; ++v) {
            int m = m0 + wr * 32 + mt * 16 + v + 8 * khalf;
#pragma unroll
            for (int nt = 0; nt < 2; ++nt) {
                int n = n0 + wc * 32 + nt * 16 + (lane & 15);
                float val = acc[mt][nt][v] + bias[n];
                if (act == 1) val = gelu_exact(val);
                if (res) val += res[(size_t)m * ldr + n];
                if (outHalf)
                    ((_Float16*)Cout)[(size_t)m * ldc + coff + n] = (_Float16)val;
                else
                    ((float*)Cout)[(size_t)m * ldc + coff + n] = val;
            }
        }
    }
}

// ---------------------------------------------------------------------------
// Flash attention, one (group, head [, q-block]) per workgroup; 128 threads.
// mode 0: dense 8x8 windows (bx = window 0..1023), N=64, 1 KV tile.
// mode 1: sparse interval-16 (bx = group 0..255, bz = q-block 0..3), N=256.
// ---------------------------------------------------------------------------
__device__ __forceinline__ int tok_of_row(int mode, int bx, int r) {
    if (mode == 0) {
        int wh = bx >> 5, ww = bx & 31;
        return ((wh * 8 + (r >> 3)) << 8) + ww * 8 + (r & 7);
    } else {
        int ih = bx >> 4, iw = bx & 15;
        int gh = r >> 4, gw = r & 15;
        return ((gh * 16 + ih) << 8) + gw * 16 + iw;
    }
}

__device__ __forceinline__ float bias_of(int mode, const float* __restrict__ pos,
                                         int head, int qr, int kc) {
    int dr, dc, span;
    if (mode == 0) { dr = (qr >> 3) - (kc >> 3); dc = (qr & 7) - (kc & 7);  span = 15; }
    else           { dr = (qr >> 4) - (kc >> 4); dc = (qr & 15) - (kc & 15); span = 31; }
    int idx = (dr + (span >> 1)) * span + (dc + (span >> 1));
    return pos[idx * 6 + head];
}

__global__ __launch_bounds__(128)
void attn_flash(const _Float16* __restrict__ qkv,
                const float* __restrict__ pos,
                _Float16* __restrict__ ctx, int mode) {
    __shared__ _Float16 Qs[64][32];
    __shared__ _Float16 Ks[64][32];
    __shared__ _Float16 Vt[32][64];      // transposed V: Vt[d][kv]
    __shared__ float    Sb[64][64];
    __shared__ _Float16 Pb[64][64];
    __shared__ float    rowM[64], rowL[64], rowA[64];

    int tid   = threadIdx.x;
    int wid   = tid >> 5;
    int lane  = tid & 31;
    int khalf = lane >> 4;
    int bx    = blockIdx.x;
    int head  = blockIdx.y;
    int qb    = mode ? blockIdx.z : 0;
    const float qscale = 0.17677669529663689f;   // 32^-0.5

    for (int i = tid; i < 64 * 32; i += 128) {
        int r = i >> 5, d = i & 31;
        int tok = tok_of_row(mode, bx, qb * 64 + r);
        float qv = (float)qkv[(size_t)tok * QKV_DIM + head * HEADD + d] * qscale;
        Qs[r][d] = (_Float16)qv;
    }
    if (tid < 64) { rowM[tid] = -INFINITY; rowL[tid] = 0.f; }
    __syncthreads();

    v8f oc0 = {}, oc1 = {};
    int ntiles = mode ? 4 : 1;

    for (int t = 0; t < ntiles; ++t) {
        for (int i = tid; i < 64 * 32; i += 128) {
            int r = i >> 5, d = i & 31;
            int tok = tok_of_row(mode, bx, t * 64 + r);
            const _Float16* base = qkv + (size_t)tok * QKV_DIM + head * HEADD + d;
            Ks[r][d] = base[DIM];
            Vt[d][r] = base[2 * DIM];
        }
        __syncthreads();

        // S = Q K^T, wave computes rows [wid*16, wid*16+16)
        union { v16h v; unsigned u[8]; } af;
        int mrow = wid * 16 + (lane & 15);
#pragma unroll
        for (int v = 0; v < 8; ++v) {
            int k = (v >> 2) * 16 + khalf * 8 + (v & 3) * 2;
            af.u[v] = *(const unsigned*)&Qs[mrow][k];
        }
#pragma unroll
        for (int nt = 0; nt < 4; ++nt) {
            union { v16h v; unsigned u[8]; } bf;
            int ncol = nt * 16 + (lane & 15);
#pragma unroll
            for (int v = 0; v < 8; ++v) {
                int k = (v >> 2) * 16 + khalf * 8 + (v & 3) * 2;
                bf.u[v] = *(const unsigned*)&Ks[ncol][k];
            }
            v8f s = {};
            s = __builtin_amdgcn_wmma_f32_16x16x32_f16(false, af.v, false, bf.v,
                                                       (short)0, s, false, false);
#pragma unroll
            for (int v = 0; v < 8; ++v) {
                int rloc = wid * 16 + v + 8 * khalf;
                int cloc = nt * 16 + (lane & 15);
                Sb[rloc][cloc] = s[v] + bias_of(mode, pos, head,
                                                qb * 64 + rloc, t * 64 + cloc);
            }
        }
        __syncthreads();

        if (tid < 64) {
            int r = tid;
            float tmax = -INFINITY;
#pragma unroll 8
            for (int c = 0; c < 64; ++c) tmax = fmaxf(tmax, Sb[r][c]);
            float nm = fmaxf(rowM[r], tmax);
            float alpha = __expf(rowM[r] - nm);
            float sum = 0.f;
#pragma unroll 8
            for (int c = 0; c < 64; ++c) {
                float e = __expf(Sb[r][c] - nm);
                sum += e;
                Pb[r][c] = (_Float16)e;
            }
            rowA[r] = alpha;
            rowL[r] = rowL[r] * alpha + sum;
            rowM[r] = nm;
        }
        __syncthreads();

#pragma unroll
        for (int v = 0; v < 8; ++v) {
            float a = rowA[wid * 16 + v + 8 * khalf];
            oc0[v] *= a;
            oc1[v] *= a;
        }
#pragma unroll
        for (int s2 = 0; s2 < 2; ++s2) {
            union { v16h v; unsigned u[8]; } pf, vf0, vf1;
            int prow = wid * 16 + (lane & 15);
#pragma unroll
            for (int v = 0; v < 8; ++v) {
                int k = s2 * 32 + (v >> 2) * 16 + khalf * 8 + (v & 3) * 2;
                pf.u[v] = *(const unsigned*)&Pb[prow][k];
            }
            int d0 = lane & 15;
#pragma unroll
            for (int v = 0; v < 8; ++v) {
                int k = s2 * 32 + (v >> 2) * 16 + khalf * 8 + (v & 3) * 2;
                vf0.u[v] = *(const unsigned*)&Vt[d0][k];
                vf1.u[v] = *(const unsigned*)&Vt[d0 + 16][k];
            }
            oc0 = __builtin_amdgcn_wmma_f32_16x16x32_f16(false, pf.v, false, vf0.v,
                                                         (short)0, oc0, false, false);
            oc1 = __builtin_amdgcn_wmma_f32_16x16x32_f16(false, pf.v, false, vf1.v,
                                                         (short)0, oc1, false, false);
        }
        __syncthreads();
    }

#pragma unroll
    for (int v = 0; v < 8; ++v) {
        int rloc = wid * 16 + v + 8 * khalf;
        float invl = 1.0f / rowL[rloc];
        int tok = tok_of_row(mode, bx, qb * 64 + rloc);
        int c = lane & 15;
        _Float16* dst = ctx + (size_t)tok * DIM + head * HEADD;
        dst[c]      = (_Float16)(oc0[v] * invl);
        dst[c + 16] = (_Float16)(oc1[v] * invl);
    }
}

// ---------------------------------------------------------------------------
// Host launch
// ---------------------------------------------------------------------------
extern "C" void kernel_launch(void* const* d_in, const int* in_sizes, int n_in,
                              void* d_out, int out_size, void* d_ws, size_t ws_size,
                              hipStream_t stream) {
    (void)in_sizes; (void)n_in; (void)out_size; (void)ws_size;

    const float* x      = (const float*)d_in[0];
    const float* g1     = (const float*)d_in[1];
    const float* be1    = (const float*)d_in[2];
    const float* w_qkv  = (const float*)d_in[3];
    const float* b_qkv  = (const float*)d_in[4];
    const float* w_proj = (const float*)d_in[5];
    const float* b_proj = (const float*)d_in[6];
    const float* pw0 = (const float*)d_in[7];  const float* pb0 = (const float*)d_in[8];
    const float* l1g = (const float*)d_in[9];  const float* l1b = (const float*)d_in[10];
    const float* pw1 = (const float*)d_in[11]; const float* pb1 = (const float*)d_in[12];
    const float* l2g = (const float*)d_in[13]; const float* l2b = (const float*)d_in[14];
    const float* pw2 = (const float*)d_in[15]; const float* pb2 = (const float*)d_in[16];
    const float* l3g = (const float*)d_in[17]; const float* l3b = (const float*)d_in[18];
    const float* pw3 = (const float*)d_in[19]; const float* pb3 = (const float*)d_in[20];
    const float* w_fuse = (const float*)d_in[21];
    const float* b_fuse = (const float*)d_in[22];
    const float* g2     = (const float*)d_in[23];
    const float* be2    = (const float*)d_in[24];
    const float* w1     = (const float*)d_in[25];
    const float* b1m    = (const float*)d_in[26];
    const float* w2     = (const float*)d_in[27];
    const float* b2m    = (const float*)d_in[28];

    char* ws = (char*)d_ws;
    size_t off = 0;
    auto alloc = [&](size_t bytes) -> void* {
        void* p = ws + off;
        off = (off + bytes + 255) & ~(size_t)255;
        return p;
    };
    _Float16* xnh    = (_Float16*)alloc((size_t)TOKENS * DIM * 2);
    _Float16* qkvh   = (_Float16*)alloc((size_t)TOKENS * QKV_DIM * 2);
    _Float16* ctxD   = (_Float16*)alloc((size_t)TOKENS * DIM * 2);
    _Float16* ctxS   = (_Float16*)alloc((size_t)TOKENS * DIM * 2);
    _Float16* concat = (_Float16*)alloc((size_t)TOKENS * 2 * DIM * 2);
    float*    x2     = (float*)   alloc((size_t)TOKENS * DIM * 4);
    _Float16* x2n    = (_Float16*)alloc((size_t)TOKENS * DIM * 2);
    _Float16* hbuf   = (_Float16*)alloc((size_t)TOKENS * HIDDEN * 2);
    _Float16* wqkvT  = (_Float16*)alloc((size_t)QKV_DIM * DIM * 2);   // 576 x 192
    _Float16* wprojT = (_Float16*)alloc((size_t)DIM * DIM * 2);       // 192 x 192
    _Float16* wfuseT = (_Float16*)alloc((size_t)DIM * 2 * DIM * 2);   // 192 x 384
    _Float16* w1T    = (_Float16*)alloc((size_t)HIDDEN * DIM * 2);    // 768 x 192
    _Float16* w2T    = (_Float16*)alloc((size_t)DIM * HIDDEN * 2);    // 192 x 768
    float*    posD   = (float*)alloc((size_t)225 * 6 * 4);
    float*    posS   = (float*)alloc((size_t)961 * 6 * 4);

    auto cvtT = [&](const float* s, _Float16* d, int K, int N) {
        convert_transpose<<<dim3((N + 15) / 16, (K + 15) / 16), dim3(16, 16),
                            0, stream>>>(s, d, K, N);
    };
    cvtT(w_qkv,  wqkvT,  DIM, QKV_DIM);
    cvtT(w_proj, wprojT, DIM, DIM);
    cvtT(w_fuse, wfuseT, 2 * DIM, DIM);
    cvtT(w1,     w1T,    DIM, HIDDEN);
    cvtT(w2,     w2T,    HIDDEN, DIM);

    posmlp_kernel<<<(225 + 961 + 127) / 128, 128, 0, stream>>>(
        pw0, pb0, l1g, l1b, pw1, pb1, l2g, l2b, pw2, pb2, l3g, l3b, pw3, pb3,
        posD, posS);

    // LN1 -> f16
    ln_kernel<<<TOKENS / 8, 256, 0, stream>>>(x, g1, be1, xnh);

    // QKV GEMM: (65536 x 192) x (192 x 576) -> f16
    gemm_wmma<<<dim3(QKV_DIM / 64, TOKENS / 128), 256, 0, stream>>>(
        xnh, DIM, wqkvT, b_qkv, qkvh, QKV_DIM, 0,
        nullptr, 0, TOKENS, QKV_DIM, DIM, 0, 1);

    // Dense windowed attention: 1024 windows x 6 heads
    attn_flash<<<dim3(1024, NHEADS, 1), 128, 0, stream>>>(qkvh, posD, ctxD, 0);
    // Sparse interval attention: 256 groups x 6 heads x 4 q-blocks
    attn_flash<<<dim3(256, NHEADS, 4), 128, 0, stream>>>(qkvh, posS, ctxS, 1);

    // Projections -> halves of the concat buffer (ldc = 384)
    gemm_wmma<<<dim3(DIM / 64, TOKENS / 128), 256, 0, stream>>>(
        ctxD, DIM, wprojT, b_proj, concat, 2 * DIM, 0,
        nullptr, 0, TOKENS, DIM, DIM, 0, 1);
    gemm_wmma<<<dim3(DIM / 64, TOKENS / 128), 256, 0, stream>>>(
        ctxS, DIM, wprojT, b_proj, concat, 2 * DIM, DIM,
        nullptr, 0, TOKENS, DIM, DIM, 0, 1);

    // Fuse 1x1 over concat channels + residual(x) -> x2 (f32)
    gemm_wmma<<<dim3(DIM / 64, TOKENS / 128), 256, 0, stream>>>(
        concat, 2 * DIM, wfuseT, b_fuse, x2, DIM, 0,
        x, DIM, TOKENS, DIM, 2 * DIM, 0, 0);

    // LN2 -> f16
    ln_kernel<<<TOKENS / 8, 256, 0, stream>>>(x2, g2, be2, x2n);

    // MLP up (GELU exact) -> f16
    gemm_wmma<<<dim3(HIDDEN / 64, TOKENS / 128), 256, 0, stream>>>(
        x2n, DIM, w1T, b1m, hbuf, HIDDEN, 0,
        nullptr, 0, TOKENS, HIDDEN, DIM, 1, 1);

    // MLP down + residual(x2) -> d_out (f32)
    gemm_wmma<<<dim3(DIM / 64, TOKENS / 128), 256, 0, stream>>>(
        hbuf, HIDDEN, w2T, b2m, d_out, DIM, 0,
        x2, DIM, TOKENS, DIM, HIDDEN, 0, 0);
}
